// MyModel_87522843560175
// MI455X (gfx1250) — matrix operand order repaired and checked by reading, not compiled, not run on hardware
//
#include <hip/hip_runtime.h>
#include <cstdint>
#include <cstddef>

// ---------------------------------------------------------------------------
// Types for WMMA fragments (gfx1250, wave32)
// ---------------------------------------------------------------------------
typedef __attribute__((ext_vector_type(16))) __bf16 v16bf;
typedef __attribute__((ext_vector_type(8)))  float  f32x8;
typedef __attribute__((ext_vector_type(4)))  float  f32x4;
typedef __attribute__((ext_vector_type(4)))  unsigned int u32x4;

union FragBF {
  v16bf v;
  u32x4 q[2];
  unsigned short us[16];
};

__device__ __forceinline__ unsigned short f32_to_bf16(float f) {
  unsigned u = __builtin_bit_cast(unsigned, f);
  u += 0x7fffu + ((u >> 16) & 1u);   // round-to-nearest-even
  return (unsigned short)(u >> 16);
}
__device__ __forceinline__ unsigned pack_bf16x2(float lo, float hi) {
  return (unsigned)f32_to_bf16(lo) | ((unsigned)f32_to_bf16(hi) << 16);
}

__device__ __forceinline__ float fast_sigmoid(float x) {
  return 1.0f / (1.0f + __expf(-x));
}
__device__ __forceinline__ float fast_tanh(float x) {
  x = fminf(fmaxf(x, -20.0f), 20.0f);
  const float e = __expf(-2.0f * x);
  return (1.0f - e) / (1.0f + e);
}

// ---------------------------------------------------------------------------
// 16-bit A-fragment load (16x32 tile, rows m0..m0+15, K = kt*32..+32) from a
// row-major bf16 matrix with row stride 512.
// ---------------------------------------------------------------------------
__device__ __forceinline__ void load_fragA(const unsigned short* __restrict__ arow,
                                           int kt, int hlf, FragBF& a) {
  const int k0 = kt * 32 + 8 * hlf;
  a.q[0] = *(const u32x4*)(arow + k0);
  a.q[1] = *(const u32x4*)(arow + k0 + 16);
}

// ---------------------------------------------------------------------------
// Weight swizzle (global): element (k,n) -> block (n>>4)*16 + (k>>5), within
// block lane=(n&15)+16*((k>>3)&1) slot=(k&7)+8*((k>>4)&1) => B fragment = contiguous 1KB per tile, and
// one N-strip (512x16) = 16 contiguous KB (ideal for LDS staging).
// ---------------------------------------------------------------------------

// Triple-gate GEMM with B fragments in LDS (strips: gate g at ldsB + g*8192).
__device__ __forceinline__ void gemm3_lds(const unsigned short* __restrict__ A, // [64][512] bf16
                                          const unsigned short* ldsB,
                                          int m0, int lane, f32x8 acc[3]) {
  const int mrow = lane & 15, hlf = lane >> 4;
  const unsigned short* arow = A + (m0 + mrow) * 512;
  const unsigned short* bl = ldsB + (lane << 4);
  for (int kt = 0; kt < 16; ++kt) {
    FragBF a;
    load_fragA(arow, kt, hlf, a);
#pragma unroll
    for (int g = 0; g < 3; ++g) {
      FragBF b;
      const unsigned short* bp = bl + g * 8192 + kt * 512;
      b.q[0] = *(const u32x4*)bp;
      b.q[1] = *(const u32x4*)(bp + 8);
      acc[g] = __builtin_amdgcn_wmma_f32_16x16x32_bf16(false, a.v, false, b.v, (short)0,
                                                       acc[g], false, false);
    }
  }
}

// Fused cell-1 GEMMs: xa = A0 @ W1-strips, ha = A1 @ U1-strips (6 WMMA / k-step).
__device__ __forceinline__ void gemm6_lds(const unsigned short* __restrict__ A0,
                                          const unsigned short* __restrict__ A1,
                                          const unsigned short* ldsB1,
                                          const unsigned short* ldsB2,
                                          int m0, int lane, f32x8 xa[3], f32x8 ha[3]) {
  const int mrow = lane & 15, hlf = lane >> 4;
  const unsigned short* a0row = A0 + (m0 + mrow) * 512;
  const unsigned short* a1row = A1 + (m0 + mrow) * 512;
  const unsigned short* b1l = ldsB1 + (lane << 4);
  const unsigned short* b2l = ldsB2 + (lane << 4);
  for (int kt = 0; kt < 16; ++kt) {
    FragBF a0, a1;
    load_fragA(a0row, kt, hlf, a0);
    load_fragA(a1row, kt, hlf, a1);
#pragma unroll
    for (int g = 0; g < 3; ++g) {
      FragBF b;
      const unsigned short* bp = b1l + g * 8192 + kt * 512;
      b.q[0] = *(const u32x4*)bp;
      b.q[1] = *(const u32x4*)(bp + 8);
      xa[g] = __builtin_amdgcn_wmma_f32_16x16x32_bf16(false, a0.v, false, b.v, (short)0,
                                                      xa[g], false, false);
      FragBF b2;
      const unsigned short* bp2 = b2l + g * 8192 + kt * 512;
      b2.q[0] = *(const u32x4*)bp2;
      b2.q[1] = *(const u32x4*)(bp2 + 8);
      ha[g] = __builtin_amdgcn_wmma_f32_16x16x32_bf16(false, a1.v, false, b2.v, (short)0,
                                                      ha[g], false, false);
    }
  }
}

// ---------------------------------------------------------------------------
// Grid-wide barrier for the persistent recurrent kernel (32 workgroups).
// ---------------------------------------------------------------------------
__device__ __forceinline__ void grid_barrier(unsigned* sync, unsigned nwg) {
  __threadfence();          // release prior global writes (agent scope)
  __syncthreads();
  if (threadIdx.x == 0) {
    unsigned* cnt = sync;
    unsigned* gen = sync + 1;
    const unsigned g = __hip_atomic_load(gen, __ATOMIC_RELAXED, __HIP_MEMORY_SCOPE_AGENT);
    const unsigned arrived =
        __hip_atomic_fetch_add(cnt, 1u, __ATOMIC_ACQ_REL, __HIP_MEMORY_SCOPE_AGENT);
    if (arrived == nwg - 1) {
      __hip_atomic_store(cnt, 0u, __ATOMIC_RELAXED, __HIP_MEMORY_SCOPE_AGENT);
      __hip_atomic_store(gen, g + 1u, __ATOMIC_RELEASE, __HIP_MEMORY_SCOPE_AGENT);
    } else {
      while (__hip_atomic_load(gen, __ATOMIC_ACQUIRE, __HIP_MEMORY_SCOPE_AGENT) == g) {
        __builtin_amdgcn_s_sleep(2);
      }
    }
  }
  __syncthreads();
  __threadfence();          // acquire: drop stale cached lines
}

// ---------------------------------------------------------------------------
// Kernel 0: convert W0,U0,W1,U1 (f32 512x1536) to bf16 in WMMA-B swizzle.
// ---------------------------------------------------------------------------
__global__ void __launch_bounds__(256) prep_weights_kernel(
    const float* __restrict__ W0, const float* __restrict__ U0,
    const float* __restrict__ W1, const float* __restrict__ U1,
    unsigned short* __restrict__ Wsw) {
  const size_t WELEM = (size_t)512 * 1536;
  const size_t idx = (size_t)blockIdx.x * blockDim.x + threadIdx.x;
  if (idx >= 4 * WELEM) return;
  const int m = (int)(idx / WELEM);
  const size_t rem = idx % WELEM;
  const int k = (int)(rem / 1536);
  const int n = (int)(rem % 1536);
  const float* srcs[4] = {W0, U0, W1, U1};
  const float v = srcs[m][rem];
  const size_t off = ((size_t)((n >> 4) * 16 + (k >> 5)) << 9) + (((n & 15) + (((k >> 3) & 1) << 4)) << 4) + ((k & 7) + (((k >> 4) & 1) << 3));
  Wsw[m * WELEM + off] = f32_to_bf16(v);
}

// ---------------------------------------------------------------------------
// Kernel 0b: zero bf16 h-state buffers and sync words (graph-replay safe).
// ---------------------------------------------------------------------------
__global__ void __launch_bounds__(256) init_state_kernel(
    unsigned short* __restrict__ hb, unsigned* __restrict__ sync) {
  const int i = blockIdx.x * blockDim.x + threadIdx.x;
  if (i < 131072) hb[i] = 0;
  if (i < 8) sync[i] = 0;
}

// ---------------------------------------------------------------------------
// Kernel 0c: streaming f32 -> bf16 conversion of x (once, bandwidth-bound).
// ---------------------------------------------------------------------------
__global__ void __launch_bounds__(256) cvt_x_kernel(const float* __restrict__ x,
                                                    unsigned short* __restrict__ xb) {
  const size_t i = ((size_t)blockIdx.x * blockDim.x + threadIdx.x) * 8;
  if (i >= (size_t)64 * 1024 * 512) return;
  const f32x4 a = *(const f32x4*)(x + i);
  const f32x4 b = *(const f32x4*)(x + i + 4);
  u32x4 o;
  o.x = pack_bf16x2(a[0], a[1]);
  o.y = pack_bf16x2(a[2], a[3]);
  o.z = pack_bf16x2(b[0], b[1]);
  o.w = pack_bf16x2(b[2], b[3]);
  *(u32x4*)(xb + i) = o;
}

// ---------------------------------------------------------------------------
// Kernel 1: xg = x @ W0 + bi0   ([65536,512] x [512,1536], bf16 WMMA, f32 acc)
// Each wave computes a 16x64 strip (4 N-tiles): one A fragment feeds 4 WMMAs.
// ---------------------------------------------------------------------------
__global__ void __launch_bounds__(256) xg_gemm_kernel(
    const unsigned short* __restrict__ xb,  // [64*1024][512] bf16
    const unsigned short* __restrict__ Wsw0,
    const float* __restrict__ b0,           // [2][1536]
    float* __restrict__ xg) {               // [64*1024][1536] f32
  const int wave = (int)((blockIdx.x * blockDim.x + threadIdx.x) >> 5);
  const int lane = threadIdx.x & 31;
  const int NTG = 24;                       // 96 N-tiles / 4 per wave
  const int total = (65536 / 16) * NTG;
  if (wave >= total) return;
  const int mt  = wave / NTG;
  const int nt0 = (wave % NTG) * 4;
  const int m0  = mt << 4;
  const int mrow = lane & 15, hlf = lane >> 4;
  const unsigned short* arow  = xb + (size_t)(m0 + mrow) * 512;
  const unsigned short* bbase = Wsw0 + (lane << 4);
  f32x8 acc[4] = {};
  for (int kt = 0; kt < 16; ++kt) {
    FragBF a;
    load_fragA(arow, kt, hlf, a);
#pragma unroll
    for (int nb = 0; nb < 4; ++nb) {
      FragBF b;
      const unsigned short* bp = bbase + ((size_t)((nt0 + nb) * 16 + kt) << 9);
      b.q[0] = *(const u32x4*)bp;
      b.q[1] = *(const u32x4*)(bp + 8);
      acc[nb] = __builtin_amdgcn_wmma_f32_16x16x32_bf16(false, a.v, false, b.v, (short)0,
                                                        acc[nb], false, false);
    }
  }
  const int c = lane & 15;
#pragma unroll
  for (int nb = 0; nb < 4; ++nb) {
    const int n = (nt0 + nb) * 16 + c;
    const float bias = b0[n];
#pragma unroll
    for (int r = 0; r < 8; ++r) {
      const int row = m0 + r + 8 * hlf;      // C layout: VGPR r -> M = r (+8 upper half)
      xg[(size_t)row * 1536 + n] = acc[nb][r] + bias;
    }
  }
}

// ---------------------------------------------------------------------------
// Kernel 2: persistent recurrence.  32 WGs x 128 thr; WG owns N-tile nt =
// blockIdx.x, its 4 waves own M-tiles 0..3.  The WG's 9 weight strips
// (U0,W1,U1 x 3 gates, 144 KB bf16) are staged in LDS once; every step then
// reads B fragments from LDS.  Each wave keeps its own f32 h0/h1 tile in
// registers for the whole sequence; cross-wave exchange uses the bf16 mirror.
// One grid barrier per timestep.
// ---------------------------------------------------------------------------
__global__ void __launch_bounds__(128) gru_recurrent_kernel(
    const float* __restrict__ xg,
    const unsigned short* __restrict__ UswU0,
    const unsigned short* __restrict__ WswW1,
    const unsigned short* __restrict__ UswU1,
    const float* __restrict__ b0, const float* __restrict__ b1,
    unsigned short* __restrict__ h0b, unsigned short* __restrict__ h1b,
    unsigned* __restrict__ sync, float* __restrict__ out) {
  extern __shared__ unsigned short lds[];   // 9 strips * 8192 ushort = 144 KB
  const int widx = threadIdx.x >> 5;        // 0..3 -> M-tile
  const int lane = threadIdx.x & 31;
  const int mt = widx, nt = blockIdx.x;     // 0..31 -> N-tile
  const int m0 = mt << 4, n0 = nt << 4;
  const int c = lane & 15, hlf = lane >> 4;

  // ---- stage this WG's weight strips into LDS (one time) ----
  {
    const unsigned short* mats[3] = {UswU0, WswW1, UswU1};
    for (int s = 0; s < 9; ++s) {
      const int mat = s / 3, gate = s % 3;
      const u32x4* src = (const u32x4*)(mats[mat] + ((size_t)((nt + gate * 32) * 16) << 9));
      u32x4* dst = (u32x4*)(lds + s * 8192);
      for (int i = threadIdx.x; i < 1024; i += 128) dst[i] = src[i];
    }
  }
  __syncthreads();

  const float br0z = b0[1536 + n0 + c];
  const float br0r = b0[1536 + 512 + n0 + c];
  const float br0h = b0[1536 + 1024 + n0 + c];
  const float bi1z = b1[n0 + c];
  const float bi1r = b1[512 + n0 + c];
  const float bi1h = b1[1024 + n0 + c];
  const float br1z = b1[1536 + n0 + c];
  const float br1r = b1[1536 + 512 + n0 + c];
  const float br1h = b1[1536 + 1024 + n0 + c];

  f32x8 h0reg = {};   // this wave's f32 h0 tile, lives in registers all 1024 steps
  f32x8 h1reg = {};

  for (int t = 0; t < 1024; ++t) {
    const int cur = t & 1, prev = cur ^ 1;

    // ---- issue xg loads early so HBM latency overlaps the GEMM ----
    float xzv[8], xrv[8], xhv[8];
#pragma unroll
    for (int r = 0; r < 8; ++r) {
      const int row = m0 + r + 8 * hlf;
      const size_t xoff = ((size_t)row * 1024 + t) * 1536 + n0 + c;
      xzv[r] = xg[xoff];
      xrv[r] = xg[xoff + 512];
      xhv[r] = xg[xoff + 1024];
    }

    // -------- Cell 0: hg0 = h0 @ U0 (B from LDS) --------
    f32x8 acc[3] = {};
    gemm3_lds(h0b + prev * 32768, lds, m0, lane, acc);
#pragma unroll
    for (int r = 0; r < 8; ++r) {
      const int row = m0 + r + 8 * hlf;
      const float z  = fast_sigmoid(xzv[r] + acc[0][r] + br0z);
      const float rr = fast_sigmoid(xrv[r] + acc[1][r] + br0r);
      const float cand = fast_tanh(xhv[r] + rr * (acc[2][r] + br0h));
      const float hn = z * h0reg[r] + (1.0f - z) * cand;
      h0reg[r] = hn;
      h0b[cur * 32768 + row * 512 + n0 + c] = f32_to_bf16(hn);
    }

    grid_barrier(sync, 32u);   // h0(t) complete & visible device-wide

    // -------- Cell 1: fused h0@W1 and h1@U1 (B from LDS) --------
    f32x8 xa[3] = {};
    f32x8 ha[3] = {};
    gemm6_lds(h0b + cur * 32768, h1b + prev * 32768,
              lds + 3 * 8192, lds + 6 * 8192, m0, lane, xa, ha);
#pragma unroll
    for (int r = 0; r < 8; ++r) {
      const int row = m0 + r + 8 * hlf;
      const float z  = fast_sigmoid(xa[0][r] + bi1z + ha[0][r] + br1z);
      const float rr = fast_sigmoid(xa[1][r] + bi1r + ha[1][r] + br1r);
      const float cand = fast_tanh(xa[2][r] + bi1h + rr * (ha[2][r] + br1h));
      const float hn = z * h1reg[r] + (1.0f - z) * cand;
      h1reg[r] = hn;
      h1b[cur * 32768 + row * 512 + n0 + c] = f32_to_bf16(hn);
      if (t == 1023) out[row * 512 + n0 + c] = hn;
    }
    // h1(t) writes become visible via next iteration's barrier fences.
  }
}

// ---------------------------------------------------------------------------
// Host launcher.  Workspace layout (bytes, 256-aligned):
//   [0)            6,291,456  : 4x bf16 swizzled weights (W0,U0,W1,U1)
//   [6,291,456)      262,144  : h0b[2] + h1b[2]  (bf16)
//   [6,553,600)          256  : sync counters
//   [7,340,032)  402,653,184  : xg = x@W0+bi0  (f32)
//   [409,993,216) 67,108,864  : xb = bf16(x)            -> total ~477 MB
// ---------------------------------------------------------------------------
extern "C" void kernel_launch(void* const* d_in, const int* in_sizes, int n_in,
                              void* d_out, int out_size, void* d_ws, size_t ws_size,
                              hipStream_t stream) {
  (void)in_sizes; (void)n_in; (void)out_size; (void)ws_size;
  const float* x  = (const float*)d_in[0];
  const float* W0 = (const float*)d_in[1];
  const float* U0 = (const float*)d_in[2];
  const float* b0 = (const float*)d_in[3];
  const float* W1 = (const float*)d_in[4];
  const float* U1 = (const float*)d_in[5];
  const float* b1 = (const float*)d_in[6];
  float* out = (float*)d_out;

  char* ws = (char*)d_ws;
  const size_t WELEM = (size_t)512 * 1536;
  unsigned short* Wsw = (unsigned short*)(ws + 0);
  unsigned short* hb  = (unsigned short*)(ws + 6291456);
  unsigned short* h0b = hb;
  unsigned short* h1b = hb + 65536;
  unsigned*       syn = (unsigned*)(ws + 6553600);
  float*          xg  = (float*)(ws + 7340032);
  unsigned short* xb  = (unsigned short*)(ws + 409993216);

  prep_weights_kernel<<<(unsigned)((4 * WELEM + 255) / 256), 256, 0, stream>>>(W0, U0, W1, U1, Wsw);
  init_state_kernel<<<512, 256, 0, stream>>>(hb, syn);
  cvt_x_kernel<<<16384, 256, 0, stream>>>(x, xb);
  xg_gemm_kernel<<<12288, 256, 0, stream>>>(xb, Wsw, b0, xg);
  gru_recurrent_kernel<<<32, 128, 9 * 8192 * sizeof(unsigned short), stream>>>(
      xg, Wsw + WELEM, Wsw + 2 * WELEM, Wsw + 3 * WELEM, b0, b1,
      h0b, h1b, syn, out);
}
